// MRGSR_29566554865686
// MI455X (gfx1250) — compile-verified, matrix operand order checked by reading, not run on hardware
//
#include <hip/hip_runtime.h>

// ---------------- problem constants (from reference) ----------------
#define N_NODES  40002
#define NP       40016          // ceil16(N_NODES)
#define ITEM_NUM 40000
#define DD       100
#define KP       128            // K=100 zero-padded for 16x16x32 bf16 WMMA
#define K3       300
#define K3P      320
#define LL       19
#define BB       512
#define BL       (BB*LL)        // 9728 (multiple of 16)
#define EE       400000
#define SQRT_D   10.0f
#define NEGBIG   2.0e32f
#define NTP      112            // ceil16(100) columns for transposed weights

typedef __attribute__((ext_vector_type(16))) __bf16       v16bf;
typedef __attribute__((ext_vector_type(8)))  float        v8f;
typedef __attribute__((ext_vector_type(8)))  unsigned int v8u;

__device__ __forceinline__ __bf16 f2bf(float f) {
  unsigned int u = __builtin_bit_cast(unsigned int, f);
  u += 0x7FFFu + ((u >> 16) & 1u);            // round-to-nearest-even
  unsigned short s = (unsigned short)(u >> 16);
  return __builtin_bit_cast(__bf16, s);
}

// ---------------- SpMM: out[rows[e]] += vals[e] * x[cols[e]] (atomic) ----------------
__global__ void spmm_atomic_kernel(const int* __restrict__ rows, const int* __restrict__ cols,
                                   const float* __restrict__ vals, const float* __restrict__ x,
                                   float* __restrict__ out) {
  long tid = (long)blockIdx.x * blockDim.x + threadIdx.x;
  long e = tid / 25;
  if (e >= EE) return;
  int c4 = (int)(tid % 25) * 4;
  int r = rows[e], c = cols[e];
  float v = vals[e];
  const float4 xv = *(const float4*)(x + (size_t)c * DD + c4);
  float* o = out + (size_t)r * DD + c4;
  atomicAdd(o + 0, v * xv.x);
  atomicAdd(o + 1, v * xv.y);
  atomicAdd(o + 2, v * xv.z);
  atomicAdd(o + 3, v * xv.w);
}

// ---------------- packing kernels (float -> zero-padded bf16 operand buffers) ----------------
__global__ void pack_a_kernel(const float* __restrict__ src, __bf16* __restrict__ dst,
                              int M, int K, int Mp, int Kp) {
  long i = (long)blockIdx.x * blockDim.x + threadIdx.x;
  if (i >= (long)Mp * Kp) return;
  int m = (int)(i / Kp), k = (int)(i % Kp);
  float v = (m < M && k < K) ? src[(size_t)m * K + k] : 0.0f;
  dst[i] = f2bf(v);
}

__global__ void pack_a_mul_kernel(const float* __restrict__ s0, const float* __restrict__ s1,
                                  __bf16* __restrict__ dst, int M, int K, int Mp, int Kp) {
  long i = (long)blockIdx.x * blockDim.x + threadIdx.x;
  if (i >= (long)Mp * Kp) return;
  int m = (int)(i / Kp), k = (int)(i % Kp);
  float v = (m < M && k < K) ? s0[(size_t)m * K + k] * s1[(size_t)m * K + k] : 0.0f;
  dst[i] = f2bf(v);
}

__global__ void pack_a_concat3_kernel(const float* __restrict__ a0, const float* __restrict__ a1,
                                      const float* __restrict__ a2, __bf16* __restrict__ dst,
                                      int M, int Mp, int Kp) {
  long i = (long)blockIdx.x * blockDim.x + threadIdx.x;
  if (i >= (long)Mp * Kp) return;
  int m = (int)(i / Kp), k = (int)(i % Kp);
  float v = 0.0f;
  if (m < M) {
    if (k < DD)            v = a0[(size_t)m * DD + k];
    else if (k < 2 * DD)   v = a1[(size_t)m * DD + (k - DD)];
    else if (k < 3 * DD)   v = a2[(size_t)m * DD + (k - 2 * DD)];
  }
  dst[i] = f2bf(v);
}

// W[K,Nc] row-major  ->  Wt[Ntp, Kp] (column-major access pattern for B fragments)
__global__ void pack_bt_kernel(const float* __restrict__ W, __bf16* __restrict__ dst,
                               int K, int Nc, int Ntp, int Kp) {
  long i = (long)blockIdx.x * blockDim.x + threadIdx.x;
  if (i >= (long)Ntp * Kp) return;
  int n = (int)(i / Kp), k = (int)(i % Kp);
  float v = (n < Nc && k < K) ? W[(size_t)k * Nc + n] : 0.0f;
  dst[i] = f2bf(v);
}

// sess_bf[b*L+l, :] = bf16(h[items[b,l], :]) zero-padded to KP
__global__ void pack_gather_sess_kernel(const float* __restrict__ h, const int* __restrict__ items,
                                        __bf16* __restrict__ dst) {
  long i = (long)blockIdx.x * blockDim.x + threadIdx.x;
  if (i >= (long)BL * KP) return;
  int row = (int)(i / KP), k = (int)(i % KP);
  float v = (k < DD) ? h[(size_t)items[row] * DD + k] : 0.0f;
  dst[i] = f2bf(v);
}

// ---------------- bf16 WMMA GEMM with TN-wide register blocking ----------------
// One wave32 computes a 16 x (16*TN) strip: the A fragment is loaded once per K-step
// and reused for TN v_wmma_f32_16x16x32_bf16 issues (TN accumulators live in VGPRs).
// tilesN must be a multiple of TN (guaranteed by operand padding).
template <int TN>
__global__ void gemm_bf16_wmma_kernel(const __bf16* __restrict__ A,   // [Mp, Kp] row-major, zero padded
                                      const __bf16* __restrict__ Bt,  // [Ntp, Kp] = W^T, zero padded
                                      const float*  __restrict__ bias,// [Nc] or nullptr
                                      float* __restrict__ C,          // [M, ldc]
                                      int M, int Nc, int Kp, int ldc, int groupsN, int act) {
  int wave = (int)(((long)blockIdx.x * blockDim.x + threadIdx.x) >> 5);
  int lane = threadIdx.x & 31;
  int gn = wave % groupsN;          // strip of TN column-tiles
  int tm = wave / groupsN;
  if (tm * 16 >= M) return;         // uniform per wave -> EXEC all-1s inside active waves

  int sub  = lane & 15;             // row (A) / col (B) within tile
  int hoff = (lane >> 4) * 8;       // K-half offset for fragments; row-half offset for C
  const int kwords = Kp >> 1;
  const unsigned int* Arow = (const unsigned int*)A + (size_t)(tm * 16 + sub) * kwords;
  const unsigned int* Bbase = (const unsigned int*)Bt + (size_t)(gn * TN * 16 + sub) * kwords;

  v8f acc[TN];
#pragma unroll
  for (int j = 0; j < TN; ++j) acc[j] = (v8f){};

  for (int k0 = 0; k0 < Kp; k0 += 32) {
    v8u ua;
#pragma unroll
    for (int i = 0; i < 8; ++i) {
      int kb = ((i < 4) ? (2 * i) : (16 + 2 * (i - 4))) + hoff + k0;
      ua[i] = Arow[kb >> 1];
    }
    v16bf a = __builtin_bit_cast(v16bf, ua);
#pragma unroll
    for (int j = 0; j < TN; ++j) {
      const unsigned int* Bcol = Bbase + (size_t)j * 16 * kwords;
      v8u ub;
#pragma unroll
      for (int i = 0; i < 8; ++i) {
        int kb = ((i < 4) ? (2 * i) : (16 + 2 * (i - 4))) + hoff + k0;
        ub[i] = Bcol[kb >> 1];
      }
      v16bf b = __builtin_bit_cast(v16bf, ub);
      acc[j] = __builtin_amdgcn_wmma_f32_16x16x32_bf16(false, a, false, b, (short)0, acc[j], false, false);
    }
  }

  int rbase = tm * 16 + hoff;
#pragma unroll
  for (int j = 0; j < TN; ++j) {
    int col = (gn * TN + j) * 16 + sub;
    if (col < Nc) {
      float bv = bias ? bias[col] : 0.0f;
#pragma unroll
      for (int v = 0; v < 8; ++v) {
        int r = rbase + v;
        if (r < M) {
          float x = acc[j][v] + bv;
          if (act == 1) x = fmaxf(x, 0.0f);
          C[(size_t)r * ldc + col] = x;
        }
      }
    }
  }
}

// ---------------- small elementwise / fusion kernels ----------------
__global__ void rowsum_div_kernel(const float* __restrict__ C, float* __restrict__ out, int M) {
  int n = blockIdx.x * blockDim.x + threadIdx.x;
  if (n >= M) return;
  float s = 0.0f;
  for (int d = 0; d < DD; ++d) s += C[(size_t)n * DD + d];
  out[n] = s * (1.0f / SQRT_D);
}

__global__ void fuse_seq_co_kernel(const float* __restrict__ emb, const float* __restrict__ in_n,
                                   const float* __restrict__ out_n, const float* __restrict__ co_n,
                                   const float* __restrict__ in_score, const float* __restrict__ out_score,
                                   const float* __restrict__ csw, const float* __restrict__ csb,
                                   const float* __restrict__ ccw, const float* __restrict__ ccb,
                                   float* __restrict__ h_seq, float* __restrict__ h_co) {
  long i = (long)blockIdx.x * blockDim.x + threadIdx.x;
  if (i >= (long)N_NODES * DD) return;
  int n = (int)(i / DD);
  float si = in_score[n], so = out_score[n];
  float mx = fmaxf(si, so);
  float ei = expf(si - mx), eo = expf(so - mx);
  float inv = 1.0f / (ei + eo);
  float nb = in_n[i] * (ei * inv) + out_n[i] * (eo * inv);
  h_seq[i] = emb[i] * csw[0] + nb * csw[1] + csb[0];
  h_co[i]  = co_n[i] * ccw[0] + emb[i] * ccw[1] + ccb[0];
}

__global__ void mean_half_kernel(const float* __restrict__ a, const float* __restrict__ nbr,
                                 float* __restrict__ out) {
  long i = (long)blockIdx.x * blockDim.x + threadIdx.x;
  if (i >= (long)N_NODES * DD) return;
  out[i] = (a[i] + nbr[i]) * 0.5f;
}

__global__ void attn_kernel(const float* __restrict__ q, const float* __restrict__ k,
                            const float* __restrict__ v, const float* __restrict__ mask,
                            float* __restrict__ sess_out) {
  __shared__ float sq[LL * DD];
  __shared__ float sk[LL * DD];
  __shared__ float sv[LL * DD];
  __shared__ float sal[LL * LL];
  int b = blockIdx.x, t = threadIdx.x;
  size_t base = (size_t)b * LL * DD;
  for (int i = t; i < LL * DD; i += blockDim.x) { sq[i] = q[base + i]; sk[i] = k[base + i]; sv[i] = v[base + i]; }
  __syncthreads();
  for (int i = t; i < LL * LL; i += blockDim.x) {
    int l = i / LL, m = i % LL;
    float s = 0.0f;
    for (int d = 0; d < DD; ++d) s += sq[l * DD + d] * sk[m * DD + d];
    s *= (1.0f / SQRT_D);
    s += (mask[b * LL + l] - 1.0f) * NEGBIG + (mask[b * LL + m] - 1.0f) * NEGBIG;
    sal[i] = s;
  }
  __syncthreads();
  if (t < LL) {
    float mx = -3.4e38f;
    for (int m = 0; m < LL; ++m) mx = fmaxf(mx, sal[t * LL + m]);
    float sum = 0.0f;
    for (int m = 0; m < LL; ++m) { float e = expf(sal[t * LL + m] - mx); sal[t * LL + m] = e; sum += e; }
    float scale = mask[b * LL + t] / sum;            // softmax(axis=2) * mask[b,l]
    for (int m = 0; m < LL; ++m) sal[t * LL + m] *= scale;
  }
  __syncthreads();
  for (int i = t; i < LL * DD; i += blockDim.x) {
    int l = i / DD, d = i % DD;
    float s = 0.0f;
    for (int m = 0; m < LL; ++m) s += sal[l * LL + m] * sv[m * DD + d];
    sess_out[base + i] = s;
  }
}

__global__ void feat_kernel(const float* __restrict__ sess_out, const float* __restrict__ gamma,
                            const float* __restrict__ beta, const float* __restrict__ mask,
                            float* __restrict__ feat) {
  long i = (long)blockIdx.x * blockDim.x + threadIdx.x;
  if (i >= (long)BL * DD) return;
  int row = (int)(i / DD);
  int l = row % LL;
  feat[i] = (sess_out[i] * gamma[l] + beta[l]) * mask[row];
}

__global__ void gather_last_kernel(const float* __restrict__ sess_out, const int* __restrict__ last_idx,
                                   float* __restrict__ last) {
  int i = blockIdx.x * blockDim.x + threadIdx.x;
  if (i >= BB * DD) return;
  int b = i / DD, d = i % DD;
  last[i] = sess_out[((size_t)b * LL + last_idx[b]) * DD + d];
}

__global__ void readout_kernel(const float* __restrict__ feat, const float* __restrict__ feat_u,
                               const float* __restrict__ feat_v, const float* __restrict__ fc_e_w,
                               const float* __restrict__ mask, const float* __restrict__ prelu_w,
                               float* __restrict__ rst) {
  __shared__ float sb[LL];
  int b = blockIdx.x, t = threadIdx.x;  // 128 threads
  if (t < LL) {
    float mk = mask[b * LL + t];
    float s = 0.0f;
    for (int d = 0; d < DD; ++d) {
      float x = feat_u[((size_t)b * LL + t) * DD + d] * mk + feat_v[(size_t)b * DD + d];
      s += (1.0f / (1.0f + expf(-x))) * fc_e_w[d];
    }
    sb[t] = s + (mk - 1.0f) * NEGBIG;
  }
  __syncthreads();
  if (t == 0) {
    float mx = -3.4e38f;
    for (int l = 0; l < LL; ++l) mx = fmaxf(mx, sb[l]);
    float sum = 0.0f;
    for (int l = 0; l < LL; ++l) { float e = expf(sb[l] - mx); sb[l] = e; sum += e; }
    float inv = 1.0f / sum;
    for (int l = 0; l < LL; ++l) sb[l] *= inv;
  }
  __syncthreads();
  if (t < DD) {
    float s = 0.0f;
    for (int l = 0; l < LL; ++l)
      s += feat[((size_t)b * LL + l) * DD + t] * sb[l] * mask[b * LL + l];
    s = (s >= 0.0f) ? s : prelu_w[t] * s;
    rst[(size_t)b * DD + t] = s;
  }
}

// ---------------- host orchestration ----------------
static inline int cdiv_l(long a, int b) { return (int)((a + b - 1) / b); }

template <int TN>
static void launch_gemm(const __bf16* A, const __bf16* Bt, const float* bias, float* C,
                        int M, int Nc, int Kp, int ldc, int act, hipStream_t stream) {
  int tilesM = (M + 15) / 16;
  int tilesN = (Nc + 15) / 16;
  int groupsN = (tilesN + TN - 1) / TN;          // operand padding guarantees exact fit
  long waves = (long)tilesM * groupsN;
  int blocks = cdiv_l(waves * 32, 256);
  gemm_bf16_wmma_kernel<TN><<<blocks, 256, 0, stream>>>(A, Bt, bias, C, M, Nc, Kp, ldc, groupsN, act);
}

extern "C" void kernel_launch(void* const* d_in, const int* in_sizes, int n_in,
                              void* d_out, int out_size, void* d_ws, size_t ws_size,
                              hipStream_t stream) {
  (void)in_sizes; (void)n_in; (void)out_size; (void)ws_size;
  const int*   items      = (const int*)d_in[0];
  const int*   last_idx   = (const int*)d_in[1];
  const float* mask       = (const float*)d_in[2];
  const float* emb        = (const float*)d_in[3];
  const int*   g_rows     = (const int*)d_in[4];
  const int*   g_cols     = (const int*)d_in[5];
  const float* g_vals     = (const float*)d_in[6];
  const float* conv_seq_w = (const float*)d_in[7];
  const float* conv_seq_b = (const float*)d_in[8];
  const float* conv_co_w  = (const float*)d_in[9];
  const float* conv_co_b  = (const float*)d_in[10];
  const float* W1_w = (const float*)d_in[11];
  const float* W1_b = (const float*)d_in[12];
  const float* W2_w = (const float*)d_in[13];
  const float* W2_b = (const float*)d_in[14];
  const float* wq   = (const float*)d_in[15];
  const float* wk   = (const float*)d_in[16];
  const float* wv   = (const float*)d_in[17];
  const float* mlp3_w   = (const float*)d_in[18];
  const float* bn_gamma = (const float*)d_in[19];
  const float* bn_beta  = (const float*)d_in[20];
  const float* fc_u_w   = (const float*)d_in[21];
  const float* fc_v_w   = (const float*)d_in[22];
  const float* fc_v_b   = (const float*)d_in[23];
  const float* fc_e_w   = (const float*)d_in[24];
  const float* prelu_w  = (const float*)d_in[25];

  // ---- carve workspace (~210 MB) ----
  char* ws = (char*)d_ws;
  size_t off = 0;
  auto carveF = [&](size_t n) -> float* { float* p = (float*)(ws + off); off += ((n * 4 + 255) / 256) * 256; return p; };
  auto carveH = [&](size_t n) -> __bf16* { __bf16* p = (__bf16*)(ws + off); off += ((n * 2 + 255) / 256) * 256; return p; };

  const size_t ND = (size_t)N_NODES * DD;
  float* in_n   = carveF(ND);
  float* out_n  = carveF(ND);
  float* co_n   = carveF(ND);
  float* nbr    = carveF(ND);
  float* h_in   = carveF(ND);
  float* h_seq  = carveF(ND);
  float* h_co   = carveF(ND);
  float* h      = carveF(ND);
  float* Ctmp   = carveF(ND);
  float* in_score  = carveF(N_NODES);
  float* out_score = carveF(N_NODES);
  float* qb       = carveF((size_t)BL * DD);
  float* kb       = carveF((size_t)BL * DD);
  float* vb       = carveF((size_t)BL * DD);
  float* sess_out = carveF((size_t)BL * DD);
  float* feat     = carveF((size_t)BL * DD);
  float* feat_u   = carveF((size_t)BL * DD);
  float* feat_v   = carveF((size_t)BB * DD);
  float* lastb    = carveF((size_t)BB * DD);
  float* rst      = carveF((size_t)BB * DD);
  __bf16* Abuf      = carveH((size_t)NP * K3P);
  __bf16* Btbuf     = carveH((size_t)NTP * K3P);
  __bf16* sess_bf   = carveH((size_t)BL * KP);
  __bf16* feat_bf   = carveH((size_t)BL * KP);
  __bf16* last_bf   = carveH((size_t)BB * KP);
  __bf16* rst_bf    = carveH((size_t)BB * KP);
  __bf16* hitems_bf = carveH((size_t)ITEM_NUM * KP);

  const int TB = 256;
  auto spmm = [&](int g, const float* x, float* out) {
    spmm_atomic_kernel<<<cdiv_l((long)EE * 25, TB), TB, 0, stream>>>(
        g_rows + (size_t)g * EE, g_cols + (size_t)g * EE, g_vals + (size_t)g * EE, x, out);
  };

  // ---- 1) neighbor aggregations (atomic SpMM) ----
  hipMemsetAsync(in_n, 0, ND * 4, stream);
  hipMemsetAsync(out_n, 0, ND * 4, stream);
  hipMemsetAsync(co_n, 0, ND * 4, stream);
  spmm(0, emb, in_n);
  spmm(1, emb, out_n);
  spmm(2, emb, co_n);

  // ---- 2) gated in/out scores: relu((emb*n) @ W + b) rowsum / sqrt(D) ----
  pack_a_mul_kernel<<<cdiv_l((long)NP * KP, TB), TB, 0, stream>>>(emb, in_n, Abuf, N_NODES, DD, NP, KP);
  pack_bt_kernel<<<cdiv_l((long)NTP * KP, TB), TB, 0, stream>>>(W1_w, Btbuf, DD, DD, NTP, KP);
  launch_gemm<7>(Abuf, Btbuf, W1_b, Ctmp, N_NODES, DD, KP, DD, /*relu*/1, stream);
  rowsum_div_kernel<<<cdiv_l(N_NODES, TB), TB, 0, stream>>>(Ctmp, in_score, N_NODES);

  pack_a_mul_kernel<<<cdiv_l((long)NP * KP, TB), TB, 0, stream>>>(emb, out_n, Abuf, N_NODES, DD, NP, KP);
  pack_bt_kernel<<<cdiv_l((long)NTP * KP, TB), TB, 0, stream>>>(W2_w, Btbuf, DD, DD, NTP, KP);
  launch_gemm<7>(Abuf, Btbuf, W2_b, Ctmp, N_NODES, DD, KP, DD, 1, stream);
  rowsum_div_kernel<<<cdiv_l(N_NODES, TB), TB, 0, stream>>>(Ctmp, out_score, N_NODES);

  fuse_seq_co_kernel<<<cdiv_l((long)ND, TB), TB, 0, stream>>>(
      emb, in_n, out_n, co_n, in_score, out_score,
      conv_seq_w, conv_seq_b, conv_co_w, conv_co_b, h_seq, h_co);

  // ---- 3) InGGNN: 2 layers of h = (h + (A3+A4) h) / 2 ----
  hipMemsetAsync(nbr, 0, ND * 4, stream);
  spmm(3, emb, nbr);
  spmm(4, emb, nbr);
  mean_half_kernel<<<cdiv_l((long)ND, TB), TB, 0, stream>>>(emb, nbr, h_in);
  hipMemsetAsync(nbr, 0, ND * 4, stream);
  spmm(3, h_in, nbr);
  spmm(4, h_in, nbr);
  mean_half_kernel<<<cdiv_l((long)ND, TB), TB, 0, stream>>>(h_in, nbr, h_in);

  // ---- 4) relation fusion: h = [h_seq|h_co|h_in] @ mlp3_w ----
  pack_a_concat3_kernel<<<cdiv_l((long)NP * K3P, TB), TB, 0, stream>>>(h_seq, h_co, h_in, Abuf, N_NODES, NP, K3P);
  pack_bt_kernel<<<cdiv_l((long)NTP * K3P, TB), TB, 0, stream>>>(mlp3_w, Btbuf, K3, DD, NTP, K3P);
  launch_gemm<7>(Abuf, Btbuf, nullptr, h, N_NODES, DD, K3P, DD, 0, stream);

  // ---- 5) session gather + self-attention ----
  pack_gather_sess_kernel<<<cdiv_l((long)BL * KP, TB), TB, 0, stream>>>(h, items, sess_bf);
  pack_bt_kernel<<<cdiv_l((long)NTP * KP, TB), TB, 0, stream>>>(wq, Btbuf, DD, DD, NTP, KP);
  launch_gemm<7>(sess_bf, Btbuf, nullptr, qb, BL, DD, KP, DD, 1, stream);
  pack_bt_kernel<<<cdiv_l((long)NTP * KP, TB), TB, 0, stream>>>(wk, Btbuf, DD, DD, NTP, KP);
  launch_gemm<7>(sess_bf, Btbuf, nullptr, kb, BL, DD, KP, DD, 1, stream);
  pack_bt_kernel<<<cdiv_l((long)NTP * KP, TB), TB, 0, stream>>>(wv, Btbuf, DD, DD, NTP, KP);
  launch_gemm<7>(sess_bf, Btbuf, nullptr, vb, BL, DD, KP, DD, 1, stream);
  attn_kernel<<<BB, TB, 0, stream>>>(qb, kb, vb, mask, sess_out);

  // ---- 6) attention readout ----
  feat_kernel<<<cdiv_l((long)BL * DD, TB), TB, 0, stream>>>(sess_out, bn_gamma, bn_beta, mask, feat);
  gather_last_kernel<<<cdiv_l((long)BB * DD, TB), TB, 0, stream>>>(sess_out, last_idx, lastb);
  pack_a_kernel<<<cdiv_l((long)BL * KP, TB), TB, 0, stream>>>(feat, feat_bf, BL, DD, BL, KP);
  pack_bt_kernel<<<cdiv_l((long)NTP * KP, TB), TB, 0, stream>>>(fc_u_w, Btbuf, DD, DD, NTP, KP);
  launch_gemm<7>(feat_bf, Btbuf, nullptr, feat_u, BL, DD, KP, DD, 0, stream);
  pack_a_kernel<<<cdiv_l((long)BB * KP, TB), TB, 0, stream>>>(lastb, last_bf, BB, DD, BB, KP);
  pack_bt_kernel<<<cdiv_l((long)NTP * KP, TB), TB, 0, stream>>>(fc_v_w, Btbuf, DD, DD, NTP, KP);
  launch_gemm<7>(last_bf, Btbuf, fc_v_b, feat_v, BB, DD, KP, DD, 0, stream);
  readout_kernel<<<BB, 128, 0, stream>>>(feat, feat_u, feat_v, fc_e_w, mask, prelu_w, rst);

  // ---- 7) final scoring: scores = rst @ h[:ITEM_NUM]^T ----
  // 2500 column tiles, TN=10 strip per wave: A (rst) fragments reused 10x per K-step.
  pack_a_kernel<<<cdiv_l((long)BB * KP, TB), TB, 0, stream>>>(rst, rst_bf, BB, DD, BB, KP);
  pack_a_kernel<<<cdiv_l((long)ITEM_NUM * KP, TB), TB, 0, stream>>>(h, hitems_bf, ITEM_NUM, DD, ITEM_NUM, KP);
  launch_gemm<10>(rst_bf, hitems_bf, nullptr, (float*)d_out, BB, ITEM_NUM, KP, ITEM_NUM, 0, stream);
}
